// LocalAttention_20315195310509
// MI455X (gfx1250) — matrix operand order repaired and checked by reading, not compiled, not run on hardware
//
#include <hip/hip_runtime.h>
#include <hip/hip_bf16.h>

// ---- problem constants (from the reference) ----
#define BB      2
#define NTOK    2048        // 8*16*16
#define TOK_T   8
#define TOK_H   16
#define TOK_W   16
#define HID     1024
#define NHEADS  16
#define HDIM    64          // HID / NHEADS
#define NQKV    (3 * HID)   // 3072

typedef _Float16 h16;
typedef __attribute__((ext_vector_type(16))) _Float16 v16h;
typedef __attribute__((ext_vector_type(8)))  _Float16 v8h;
typedef __attribute__((ext_vector_type(4)))  _Float16 v4h;
typedef __attribute__((ext_vector_type(4)))  float    v4f;
typedef __attribute__((ext_vector_type(8)))  float    v8f;

// ---------------------------------------------------------------------------
// Kernel 1: f32 -> f16 conversion, vectorized (b128 load, b64 store).
// All element counts here are multiples of 4.
// ---------------------------------------------------------------------------
__global__ __launch_bounds__(256)
void cvt_f32_f16(const float* __restrict__ in, h16* __restrict__ out, int n4)
{
    int i = blockIdx.x * blockDim.x + threadIdx.x;
    int stride = gridDim.x * blockDim.x;
    for (; i < n4; i += stride) {
        v4f f = *(const v4f*)(in + (size_t)i * 4);
        v4h h;
        h[0] = (h16)f[0]; h[1] = (h16)f[1];
        h[2] = (h16)f[2]; h[3] = (h16)f[3];
        *(v4h*)(out + (size_t)i * 4) = h;
    }
}

// ---------------------------------------------------------------------------
// Kernel 2/4: C[M x Nout](f32) = A[M x K](f16, row-major) @ B[Nout x K](f16)^T
// One wave computes a 32(M) x 64(N) tile: 2 A-fragments x 4 B-fragments
// -> 8 WMMAs per K-step; each B fragment reused 2x, each A fragment 4x
// (~1.5 b128 loads per WMMA). Block = 4 waves covering 32 x 256.
//
// Fragment layouts per CDNA5 ISA (16-bit A 16x32, B 32x16, f32 C 16x16):
//   A: lane = {row = lane&15, half = lane>>4}; vector elems 0..7  = K half*8+0..7
//                                              vector elems 8..15 = K 16+half*8+0..7
//   B: lane = {col = lane&15, half = lane>>4}; vector elems 0..7 = K half*8+0..7, 8..15 = K 16+half*8+0..7
//   C: elem r of v8f -> M = r + 8*half, N = lane&15
// ---------------------------------------------------------------------------
__device__ __forceinline__ v16h load_a_frag(const h16* Arow, int kk, int half)
{
    v8h lo = *(const v8h*)(Arow + kk + half * 8);
    v8h hi = *(const v8h*)(Arow + kk + 16 + half * 8);
    return __builtin_shufflevector(lo, hi,
             0,1,2,3,4,5,6,7,8,9,10,11,12,13,14,15);
}

__global__ __launch_bounds__(128)
void gemm_f16_wmma(const h16* __restrict__ A, const h16* __restrict__ B,
                   float* __restrict__ C, int M, int Nout, int K)
{
    const int lane = threadIdx.x & 31;
    const int wave = threadIdx.x >> 5;                 // 0..3
    const int m0   = blockIdx.y * 32;
    const int n0   = blockIdx.x * 256 + wave * 64;
    if (m0 >= M || n0 >= Nout) return;

    const int row  = lane & 15;                        // A row / B column within tile
    const int half = lane >> 4;

    v8f acc[2][4] = {};

    const h16* Arow0 = A + (size_t)(m0 + row) * K;
    const h16* Arow1 = A + (size_t)(m0 + 16 + row) * K;

    for (int kk = 0; kk < K; kk += 32) {
        // prefetch the streaming A rows a few K-steps ahead (speculative,
        // lowers to global_prefetch_b8; OOB prefetches are dropped)
        __builtin_prefetch(Arow0 + kk + 256, 0, 3);
        __builtin_prefetch(Arow1 + kk + 256, 0, 3);

        v16h a0 = load_a_frag(Arow0, kk, half);
        v16h a1 = load_a_frag(Arow1, kk, half);
#pragma unroll
        for (int t = 0; t < 4; ++t) {
            const h16* Brow = B + (size_t)(n0 + t * 16 + row) * K + kk + half * 8;
            v16h b = __builtin_shufflevector(*(const v8h*)Brow, *(const v8h*)(Brow + 16),
                       0,1,2,3,4,5,6,7,8,9,10,11,12,13,14,15);   // K half*8+0..7, 16+half*8+0..7
            acc[0][t] = __builtin_amdgcn_wmma_f32_16x16x32_f16(
                            false, a0, false, b, (short)0, acc[0][t], false, false);
            acc[1][t] = __builtin_amdgcn_wmma_f32_16x16x32_f16(
                            false, a1, false, b, (short)0, acc[1][t], false, false);
        }
    }

#pragma unroll
    for (int s = 0; s < 2; ++s) {
#pragma unroll
        for (int t = 0; t < 4; ++t) {
            float* Cp = C + (size_t)(m0 + s * 16 + half * 8) * Nout + n0 + t * 16 + row;
#pragma unroll
            for (int r = 0; r < 8; ++r)
                Cp[(size_t)r * Nout] = acc[s][t][r];   // M = s*16 + half*8 + r
        }
    }
}

// ---------------------------------------------------------------------------
// Kernel 3: local 3x3x3 attention. One wave32 per (b, token, head).
// Lane l owns output dims l and l+32 of the 64-dim head.
// Zero-padded OOB neighbors contribute score 0 to the softmax (matching the
// reference, which pads k with zeros) and nothing to the value sum.
// qkv layout per row (b*N+n): [0,1024)=q  [1024,2048)=k  [2048,3072)=v
// ---------------------------------------------------------------------------
__global__ __launch_bounds__(256)
void local_attn(const float* __restrict__ qkv, h16* __restrict__ attn_out)
{
    const int lane = threadIdx.x & 31;
    const int wid  = blockIdx.x * (blockDim.x >> 5) + (threadIdx.x >> 5);

    const int h = wid & (NHEADS - 1);
    const int n = (wid >> 4) & (NTOK - 1);
    const int b = wid >> 15;                           // NTOK*NHEADS = 2^15

    const int tt = n >> 8;
    const int hh = (n >> 4) & 15;
    const int ww = n & 15;

    const size_t rowq = (size_t)(b * NTOK + n);
    const float* qp = qkv + rowq * NQKV + h * HDIM;
    const float qv0 = qp[lane];
    const float qv1 = qp[lane + 32];

    // ---- pass 1: 27 scores; lane i ends up holding score i ----
    float mysc = -INFINITY;
    for (int i = 0; i < 27; ++i) {
        const int dt = i / 9 - 1;
        const int r3 = i - (i / 9) * 9;
        const int dh = r3 / 3 - 1;
        const int dw = r3 - (r3 / 3) * 3 - 1;
        const int t2 = tt + dt, h2 = hh + dh, w2 = ww + dw;
        const bool valid = ((unsigned)t2 < (unsigned)TOK_T) &
                           ((unsigned)h2 < (unsigned)TOK_H) &
                           ((unsigned)w2 < (unsigned)TOK_W);
        float s = 0.0f;
        if (valid) {                                   // uniform branch
            const int n2 = (t2 << 8) | (h2 << 4) | w2;
            const float* kp = qkv + (size_t)(b * NTOK + n2) * NQKV + HID + h * HDIM;
            s = qv0 * kp[lane] + qv1 * kp[lane + 32];
            s += __shfl_xor(s, 16);
            s += __shfl_xor(s, 8);
            s += __shfl_xor(s, 4);
            s += __shfl_xor(s, 2);
            s += __shfl_xor(s, 1);
        }
        if (lane == i) mysc = s;
    }

    // ---- softmax over 27 scores held across lanes 0..26 ----
    const float scale = 0.125f;                        // 64^-0.5
    float mx = mysc;
    mx = fmaxf(mx, __shfl_xor(mx, 16));
    mx = fmaxf(mx, __shfl_xor(mx, 8));
    mx = fmaxf(mx, __shfl_xor(mx, 4));
    mx = fmaxf(mx, __shfl_xor(mx, 2));
    mx = fmaxf(mx, __shfl_xor(mx, 1));
    float e = (lane < 27) ? __expf((mysc - mx) * scale) : 0.0f;
    float ds = e;
    ds += __shfl_xor(ds, 16);
    ds += __shfl_xor(ds, 8);
    ds += __shfl_xor(ds, 4);
    ds += __shfl_xor(ds, 2);
    ds += __shfl_xor(ds, 1);
    const float wgt = e / ds;

    // ---- pass 2: weighted value sum ----
    float a0 = 0.0f, a1 = 0.0f;
    for (int i = 0; i < 27; ++i) {
        const int dt = i / 9 - 1;
        const int r3 = i - (i / 9) * 9;
        const int dh = r3 / 3 - 1;
        const int dw = r3 - (r3 / 3) * 3 - 1;
        const int t2 = tt + dt, h2 = hh + dh, w2 = ww + dw;
        const bool valid = ((unsigned)t2 < (unsigned)TOK_T) &
                           ((unsigned)h2 < (unsigned)TOK_H) &
                           ((unsigned)w2 < (unsigned)TOK_W);
        if (valid) {                                   // uniform branch
            const float wi = __shfl(wgt, i);
            const int n2 = (t2 << 8) | (h2 << 4) | w2;
            const float* vp = qkv + (size_t)(b * NTOK + n2) * NQKV + 2 * HID + h * HDIM;
            a0 += wi * vp[lane];
            a1 += wi * vp[lane + 32];
        }
    }

    h16* op = attn_out + rowq * HID + h * HDIM;
    op[lane]      = (h16)a0;
    op[lane + 32] = (h16)a1;
}

// ---------------------------------------------------------------------------
extern "C" void kernel_launch(void* const* d_in, const int* in_sizes, int n_in,
                              void* d_out, int out_size, void* d_ws, size_t ws_size,
                              hipStream_t stream)
{
    (void)in_sizes; (void)n_in; (void)out_size; (void)ws_size;

    const float* x     = (const float*)d_in[0];   // (2, 2048, 1024)
    const float* w_qkv = (const float*)d_in[1];   // (3072, 1024)
    const float* w_out = (const float*)d_in[2];   // (1024, 1024)
    float*       out   = (float*)d_out;           // (2, 2048, 1024)

    const int M = BB * NTOK;   // 4096
    const int K = HID;         // 1024

    // workspace carve-out (256B aligned), ~75.5 MB total
    char*  ws  = (char*)d_ws;
    size_t off = 0;
    auto take = [&](size_t bytes) {
        char* p = ws + off;
        off += (bytes + 255) & ~(size_t)255;
        return p;
    };
    h16*   xh   = (h16*)  take((size_t)M * K * sizeof(h16));        // 8 MB
    h16*   wqh  = (h16*)  take((size_t)NQKV * K * sizeof(h16));     // 6 MB
    h16*   woh  = (h16*)  take((size_t)HID * K * sizeof(h16));      // 2 MB
    float* qkv  = (float*)take((size_t)M * NQKV * sizeof(float));   // 48 MB
    h16*   atth = (h16*)  take((size_t)M * HID * sizeof(h16));      // 8 MB

    // 1) convert inputs to f16 (vectorized x4)
    cvt_f32_f16<<<2048, 256, 0, stream>>>(x,     xh,  (M * K) / 4);
    cvt_f32_f16<<<2048, 256, 0, stream>>>(w_qkv, wqh, (NQKV * K) / 4);
    cvt_f32_f16<<<1024, 256, 0, stream>>>(w_out, woh, (HID * K) / 4);

    // 2) qkv = x @ w_qkv^T   (4096 x 3072, K=1024)
    gemm_f16_wmma<<<dim3(NQKV / 256, M / 32), 128, 0, stream>>>(
        xh, wqh, qkv, M, NQKV, K);

    // 3) local attention -> f16 (4096 x 1024); one wave per (b, n, head)
    //    waves = 2*2048*16 = 65536; 8 waves per 256-thread block
    local_attn<<<(BB * NTOK * NHEADS) / 8, 256, 0, stream>>>(qkv, atth);

    // 4) out = attn @ w_out^T  (4096 x 1024, K=1024)
    gemm_f16_wmma<<<dim3(HID / 256, M / 32), 128, 0, stream>>>(
        atth, woh, out, M, HID, K);
}
